// EGKN_2740189135780
// MI455X (gfx1250) — matrix operand order repaired and checked by reading, not compiled, not run on hardware
//
#include <hip/hip_runtime.h>
#include <hip/hip_bf16.h>

// ---------------- problem constants ----------------
#define N_NODES 10000
#define N_EDGES 300000
#define KIN     6
#define W       32
#define KW      256      // kernel MLP hidden width
#define KHID    128      // KW/2
#define KCOLS   1024     // W*W, per-edge kernel matrix size
#define DEPTH   4

typedef __bf16 bf16_t;
typedef __attribute__((ext_vector_type(16))) __bf16 v16bf;
typedef __attribute__((ext_vector_type(8)))  float  v8f;

#define LDS_STRIDE 264   // 16x padded-K bf16 tile; 528B row stride -> bank-conflict-free column reads

// ======================================================================
// Pack ker2_w [128,256] and ker3_w [256,1024] (fp32) into bf16 WMMA
// B-fragment order: frag index (kblk, ntile), then [lane][elem] so each
// lane's 16 bf16 (32B) are contiguous. B layout: K = kblk*32 + lanehalf*16 + e,
// N = ntile*16 + lane%16.
// ======================================================================
__global__ __launch_bounds__(256) void pack_weights(
    const float* __restrict__ w2, const float* __restrict__ w3,
    bf16_t* __restrict__ w2p, bf16_t* __restrict__ w3p)
{
    const int idx = blockIdx.x * 256 + threadIdx.x;
    if (idx < 4 * 16 * 512) {                 // stage-2 B: 4 kblks x 16 ntiles
        const int f = idx >> 9, li = idx & 511;
        const int lane = li >> 4, e = li & 15;
        const int kb = f >> 4, nt = f & 15;
        const int K = kb * 32 + (lane >> 4) * 16 + e;
        const int C = nt * 16 + (lane & 15);
        w2p[idx] = (bf16_t)w2[K * KW + C];
    } else if (idx < 4 * 16 * 512 + 8 * 64 * 512) {   // stage-3 B: 8 kblks x 64 ntiles
        const int j = idx - 4 * 16 * 512;
        const int f = j >> 9, li = j & 511;
        const int lane = li >> 4, e = li & 15;
        const int kb = f >> 6, nt = f & 63;
        const int K = kb * 32 + (lane >> 4) * 16 + e;
        const int C = nt * 16 + (lane & 15);
        w3p[j] = (bf16_t)w3[K * KCOLS + C];
    }
}

// ======================================================================
// Per-edge kernel MLP, fully fused:
//   k1 = relu(ea @ W1 + b1)            (VALU, built directly into A frags)
//   k2 = relu(k1 @ W2 + b2)            (WMMA bf16, 16 ntiles x 4 ksteps)
//   k  =       k2 @ W3 + b3            (WMMA bf16, 64 ntiles x 8 ksteps)
// One wave = 16 edges. Stage-2 D -> stage-3 A transpose goes through a
// padded LDS tile. Output k stored row-major bf16 [E][1024].
// ======================================================================
__global__ __launch_bounds__(128) void edge_kernel_mlp(
    const float* __restrict__ ea,
    const float* __restrict__ w1, const float* __restrict__ b1,
    const float* __restrict__ b2, const float* __restrict__ b3,
    const bf16_t* __restrict__ w2p, const bf16_t* __restrict__ w3p,
    bf16_t* __restrict__ kbuf)
{
    __shared__ bf16_t k2lds[4 * 16 * LDS_STRIDE];   // 4 waves x 16 rows x padded 256

    const int wave = threadIdx.x >> 5;
    const int lane = threadIdx.x & 31;
    const int m    = lane & 15;     // row (edge within tile)
    const int kh   = lane >> 4;     // lane half selects K sub-block

    const long tile  = (long)blockIdx.x * 4 + wave;
    const long ebase = tile * 16;
    long erow = ebase + m;
    if (erow >= N_EDGES) erow = N_EDGES - 1;   // clamp; padded rows never stored

    const float e0 = ea[erow * KIN + 0], e1 = ea[erow * KIN + 1],
                e2 = ea[erow * KIN + 2], e3 = ea[erow * KIN + 3],
                e4 = ea[erow * KIN + 4], e5 = ea[erow * KIN + 5];

    // ---- stage 1: build stage-2 A fragments (16x128 bf16 = 4 k-blocks) ----
    // A layout: elem t -> K = kh*16 + t
    v16bf a2[4];
#pragma unroll
    for (int kb = 0; kb < 4; ++kb) {
#pragma unroll
        for (int t = 0; t < 16; ++t) {
            const int K = kb * 32 + kh * 16 + t;
            float acc = b1[K];
            acc += e0 * w1[0 * KHID + K] + e1 * w1[1 * KHID + K]
                 + e2 * w1[2 * KHID + K] + e3 * w1[3 * KHID + K]
                 + e4 * w1[4 * KHID + K] + e5 * w1[5 * KHID + K];
            a2[kb][t] = (bf16_t)fmaxf(acc, 0.f);
        }
    }

    bf16_t* myk2 = &k2lds[wave * 16 * LDS_STRIDE];

    // ---- stage 2: [16,128] @ [128,256] ----
    for (int nt = 0; nt < 16; ++nt) {
        const float bias = b2[nt * 16 + m];
        v8f acc;
#pragma unroll
        for (int r = 0; r < 8; ++r) acc[r] = bias;
#pragma unroll
        for (int kb = 0; kb < 4; ++kb) {
            const v16bf bfrag = *(const v16bf*)(w2p + ((size_t)(kb * 16 + nt) * 32 + lane) * 16);
            acc = __builtin_amdgcn_wmma_f32_16x16x32_bf16(
                      false, a2[kb], false, bfrag, (short)0, acc, false, false);
        }
        // D layout: M = r + 8*kh, N = m ; write relu'd bf16 into LDS tile
        const int col = nt * 16 + m;
#pragma unroll
        for (int r = 0; r < 8; ++r)
            myk2[(r + 8 * kh) * LDS_STRIDE + col] = (bf16_t)fmaxf(acc[r], 0.f);
    }
    __syncthreads();

    // ---- transpose to stage-3 A fragments (16x256 bf16 = 8 k-blocks) ----
    v16bf a3[8];
#pragma unroll
    for (int kb = 0; kb < 8; ++kb) {
#pragma unroll
        for (int t = 0; t < 16; ++t) {
            const int K = kb * 32 + kh * 16 + t;
            a3[kb][t] = myk2[m * LDS_STRIDE + K];
        }
    }

    // ---- stage 3: [16,256] @ [256,1024] -> k rows ----
    for (int nt = 0; nt < 64; ++nt) {
        if (nt + 1 < 64)   // warm L0/WGP$ for the next ntile's fragments
            __builtin_prefetch(w3p + ((size_t)(nt + 1) * 32 + lane) * 16, 0, 3);
        const float bias = b3[nt * 16 + m];
        v8f acc;
#pragma unroll
        for (int r = 0; r < 8; ++r) acc[r] = bias;
#pragma unroll
        for (int kb = 0; kb < 8; ++kb) {
            const v16bf bfrag = *(const v16bf*)(w3p + ((size_t)(kb * 64 + nt) * 32 + lane) * 16);
            acc = __builtin_amdgcn_wmma_f32_16x16x32_bf16(
                      false, a3[kb], false, bfrag, (short)0, acc, false, false);
        }
        const int col = nt * 16 + m;
#pragma unroll
        for (int r = 0; r < 8; ++r) {
            const long e = ebase + r + 8 * kh;
            if (e < N_EDGES) kbuf[e * KCOLS + col] = (bf16_t)acc[r];
        }
    }
}

// ======================================================================
// Node init: h = x@fc1+b, zero accumulators, copy coords, zero cnt.
// ======================================================================
__global__ __launch_bounds__(256) void init_nodes(
    const float* __restrict__ x, const float* __restrict__ fc1w,
    const float* __restrict__ fc1b, const float* __restrict__ ci,
    float* __restrict__ h, float* __restrict__ agg,
    float* __restrict__ coord, float* __restrict__ cacc, float* __restrict__ cnt)
{
    const int idx = blockIdx.x * 256 + threadIdx.x;
    if (idx >= N_NODES * W) return;
    const int n = idx >> 5, f = idx & 31;
    float a = fc1b[f];
#pragma unroll
    for (int c = 0; c < 3; ++c) a += x[n * 3 + c] * fc1w[c * W + f];
    h[idx]   = a;
    agg[idx] = 0.f;
    if (f < 3) { coord[n * 3 + f] = ci[n * 3 + f]; cacc[n * 3 + f] = 0.f; }
    if (f == 0) cnt[n] = 0.f;
}

__global__ __launch_bounds__(256) void count_edges(const int* __restrict__ ei,
                                                   float* __restrict__ cnt)
{
    const int e = blockIdx.x * 256 + threadIdx.x;
    if (e < N_EDGES) atomicAdd(&cnt[ei[e]], 1.f);
}

// ======================================================================
// One depth-iteration, edge side: m = k_e . h[col]; scatter m into agg;
// coord weight MLP; scatter equivariant update. One edge per wave,
// lane = feature. k row read = 64B/lane contiguous bf16 (coalesced 2KB).
// ======================================================================
__global__ __launch_bounds__(256) void layer_edges(
    const bf16_t* __restrict__ kbuf, const int* __restrict__ ei,
    const float* __restrict__ h, const float* __restrict__ coord,
    const float* __restrict__ cm1w, const float* __restrict__ cm1b,
    const float* __restrict__ cm2w, const float* __restrict__ cm2b,
    float* __restrict__ agg, float* __restrict__ cacc)
{
    const int wave = threadIdx.x >> 5, lane = threadIdx.x & 31;
    const long e = (long)blockIdx.x * 8 + wave;
    if (e >= N_EDGES) return;
    const int row = ei[e], col = ei[N_EDGES + e];

    const float hv = h[(long)col * W + lane];
    const bf16_t* kr = kbuf + e * KCOLS + lane * W;
    const v16bf k0 = *(const v16bf*)kr;
    const v16bf k1 = *(const v16bf*)(kr + 16);

    float mval = 0.f;
#pragma unroll
    for (int j = 0; j < 16; ++j) mval += (float)k0[j] * __shfl(hv, j, 32);
#pragma unroll
    for (int j = 0; j < 16; ++j) mval += (float)k1[j] * __shfl(hv, 16 + j, 32);

    atomicAdd(&agg[(long)row * W + lane], mval);

    // we = relu(m @ cm1 + b) @ cm2 + b
    float t = cm1b[lane];
#pragma unroll
    for (int i = 0; i < 32; ++i) t += __shfl(mval, i, 32) * cm1w[i * W + lane];
    float p = fmaxf(t, 0.f) * cm2w[lane];
#pragma unroll
    for (int off = 16; off > 0; off >>= 1) p += __shfl_xor(p, off, 32);
    const float we = p + cm2b[0];

    if (lane < 3) {
        const float tr = (coord[row * 3 + lane] - coord[col * 3 + lane]) * we;
        atomicAdd(&cacc[row * 3 + lane], tr);
    }
}

__global__ __launch_bounds__(256) void node_update(
    float* __restrict__ h, float* __restrict__ agg,
    float* __restrict__ coord, float* __restrict__ cacc,
    const float* __restrict__ cnt)
{
    const int idx = blockIdx.x * 256 + threadIdx.x;
    if (idx >= N_NODES * W) return;
    const int n = idx >> 5, f = idx & 31;
    const float c = fmaxf(cnt[n], 1.f);
    h[idx]   = fmaxf(h[idx] + agg[idx] / c, 0.f);
    agg[idx] = 0.f;
    if (f < 3) { coord[n * 3 + f] += cacc[n * 3 + f] / c; cacc[n * 3 + f] = 0.f; }
}

// ======================================================================
// out = relu(h @ fc2a + b) @ fc2b + b ; one node per wave.
// ======================================================================
__global__ __launch_bounds__(256) void final_out(
    const float* __restrict__ h,
    const float* __restrict__ f2aw, const float* __restrict__ f2ab,
    const float* __restrict__ f2bw, const float* __restrict__ f2bb,
    float* __restrict__ dout)
{
    const int wave = threadIdx.x >> 5, lane = threadIdx.x & 31;
    const long n = (long)blockIdx.x * 8 + wave;
    if (n >= N_NODES) return;
    const float hv = h[n * W + lane];
    float t0 = f2ab[lane], t1 = f2ab[32 + lane];
#pragma unroll
    for (int i = 0; i < 32; ++i) {
        const float hi = __shfl(hv, i, 32);
        t0 += hi * f2aw[i * 64 + lane];
        t1 += hi * f2aw[i * 64 + 32 + lane];
    }
    float p = fmaxf(t0, 0.f) * f2bw[lane] + fmaxf(t1, 0.f) * f2bw[32 + lane];
#pragma unroll
    for (int off = 16; off > 0; off >>= 1) p += __shfl_xor(p, off, 32);
    if (lane == 0) dout[n] = p + f2bb[0];
}

__global__ __launch_bounds__(256) void copy_coords(const float* __restrict__ coord,
                                                   float* __restrict__ dout)
{
    const int i = blockIdx.x * 256 + threadIdx.x;
    if (i < N_NODES * 3) dout[N_NODES + i] = coord[i];
}

// ======================================================================
extern "C" void kernel_launch(void* const* d_in, const int* in_sizes, int n_in,
                              void* d_out, int out_size, void* d_ws, size_t ws_size,
                              hipStream_t stream)
{
    const float* x    = (const float*)d_in[0];
    const int*   ei   = (const int*)  d_in[1];
    const float* ea   = (const float*)d_in[2];
    const float* ci   = (const float*)d_in[3];
    const float* fc1w = (const float*)d_in[4];
    const float* fc1b = (const float*)d_in[5];
    const float* k1w  = (const float*)d_in[6];
    const float* k1b  = (const float*)d_in[7];
    const float* k2w  = (const float*)d_in[8];
    const float* k2b  = (const float*)d_in[9];
    const float* k3w  = (const float*)d_in[10];
    const float* k3b  = (const float*)d_in[11];
    const float* cm1w = (const float*)d_in[12];
    const float* cm1b = (const float*)d_in[13];
    const float* cm2w = (const float*)d_in[14];
    const float* cm2b = (const float*)d_in[15];
    const float* f2aw = (const float*)d_in[16];
    const float* f2ab = (const float*)d_in[17];
    const float* f2bw = (const float*)d_in[18];
    const float* f2bb = (const float*)d_in[19];
    float* out = (float*)d_out;

    char* ws = (char*)d_ws;
    size_t off = 0;
    auto take = [&](size_t bytes) -> void* {
        void* p = ws + off;
        off = (off + bytes + 255) & ~(size_t)255;
        return p;
    };
    bf16_t* kbuf = (bf16_t*)take((size_t)N_EDGES * KCOLS * sizeof(bf16_t)); // 600 MB
    float*  h    = (float*) take((size_t)N_NODES * W * sizeof(float));
    float*  agg  = (float*) take((size_t)N_NODES * W * sizeof(float));
    float*  crd  = (float*) take((size_t)N_NODES * 3 * sizeof(float));
    float*  cacc = (float*) take((size_t)N_NODES * 3 * sizeof(float));
    float*  cnt  = (float*) take((size_t)N_NODES * sizeof(float));
    bf16_t* w2p  = (bf16_t*)take((size_t)4 * 16 * 512 * sizeof(bf16_t));
    bf16_t* w3p  = (bf16_t*)take((size_t)8 * 64 * 512 * sizeof(bf16_t));

    // 1) pack GEMM weights to bf16 fragments
    pack_weights<<<(4 * 16 * 512 + 8 * 64 * 512 + 255) / 256, 256, 0, stream>>>(
        k2w, k3w, w2p, w3p);

    // 2) node init (h, coord, zero accumulators)
    init_nodes<<<(N_NODES * W + 255) / 256, 256, 0, stream>>>(
        x, fc1w, fc1b, ci, h, agg, crd, cacc, cnt);

    // 3) degree counts
    count_edges<<<(N_EDGES + 255) / 256, 256, 0, stream>>>(ei, cnt);

    // 4) the big one: per-edge kernel MLP -> k [E,1024] bf16 (WMMA)
    {
        const int tiles = (N_EDGES + 15) / 16;          // 18750
        const int blocks = (tiles + 3) / 4;             // 4 waves/block
        edge_kernel_mlp<<<blocks, 128, 0, stream>>>(
            ea, k1w, k1b, k2b, k3b, w2p, w3p, kbuf);
    }

    // 5) depth loop (k stream is the bandwidth floor: 600MB x 4)
    for (int d = 0; d < DEPTH; ++d) {
        layer_edges<<<(N_EDGES + 7) / 8, 256, 0, stream>>>(
            kbuf, ei, h, crd, cm1w, cm1b, cm2w, cm2b, agg, cacc);
        node_update<<<(N_NODES * W + 255) / 256, 256, 0, stream>>>(
            h, agg, crd, cacc, cnt);
    }

    // 6) outputs: out [N] then coord [N,3] flat
    final_out<<<(N_NODES + 7) / 8, 256, 0, stream>>>(h, f2aw, f2ab, f2bw, f2bb, out);
    copy_coords<<<(N_NODES * 3 + 255) / 256, 256, 0, stream>>>(crd, out);
}
